// SequenceCrossEntropyLoss_48911087566954
// MI455X (gfx1250) — matrix operand order repaired and checked by reading, compiled, not run on hardware
//
#include <hip/hip_runtime.h>
#include <math.h>

typedef __attribute__((ext_vector_type(16))) _Float16 v16h;
typedef __attribute__((ext_vector_type(4)))  _Float16 v4h;
typedef __attribute__((ext_vector_type(8)))  float    v8f;
typedef __attribute__((ext_vector_type(4)))  float    v4f;

#define EPSN 1e-8f
#define BB   32      // batch
#define SS   20      // seq len
#define DD   128     // feature dim (4 x K=32 WMMA steps)
#define NROWS (BB * SS)          // 640 packed pred rows
#define NTILE (NROWS / 16)       // 40 A-tiles, no padding

// ---------------------------------------------------------------------------
// Kernel 1: normalize all 640 pred rows -> f16, packed (no padding).
// One wave per row.
// ---------------------------------------------------------------------------
__global__ void __launch_bounds__(256)
norm_pred_kernel(const float* __restrict__ pred, _Float16* __restrict__ pred_h) {
    int row  = (blockIdx.x * blockDim.x + threadIdx.x) >> 5;
    int lane = threadIdx.x & 31;
    if (row >= NROWS) return;

    const float* rp = pred + (size_t)row * DD + lane * 4;
    float x0 = rp[0], x1 = rp[1], x2 = rp[2], x3 = rp[3];
    float ssq = x0 * x0 + x1 * x1 + x2 * x2 + x3 * x3;
    #pragma unroll
    for (int off = 16; off > 0; off >>= 1)
        ssq += __shfl_xor(ssq, off, 32);
    float inv = 1.0f / fmaxf(sqrtf(ssq), EPSN);

    v4h o;
    o[0] = (_Float16)(x0 * inv);
    o[1] = (_Float16)(x1 * inv);
    o[2] = (_Float16)(x2 * inv);
    o[3] = (_Float16)(x3 * inv);
    *reinterpret_cast<v4h*>(pred_h + (size_t)row * DD + lane * 4) = o;
}

// ---------------------------------------------------------------------------
// Kernel 2: fused song-normalize + WMMA GEMM + seq-max, zero padding waste.
// One wave per 16-song tile. A = packed pred rows (M = flat b*20+s),
// B = song columns. 40 fully-unrolled A-tiles x 4 K-steps = 160 WMMAs/wave.
// Each 16-row tile spans <=2 batches; per (tile, lane-half, batch) the valid
// accumulator elements form a compile-time contiguous range, so the masked
// max folds to a constant v_max chain. A running register carries the open
// batch's partial max; on close: one shfl_xor(16) merge + one 16-lane store.
// ---------------------------------------------------------------------------
__global__ void __launch_bounds__(256)
simmax_kernel(const float* __restrict__ song,
              const _Float16* __restrict__ pred_h,
              const float* __restrict__ x_inv,
              float* __restrict__ sim, int N) {
    int wave = blockIdx.x * 8 + (threadIdx.x >> 5);
    int lane = threadIdx.x & 31;
    int ntiles = N >> 4;
    if (wave >= ntiles) return;           // wave-uniform: EXEC stays all-1s
    int srow = wave << 4;

    // ---- B fragments (songs, 32x16 KxN layout), fused normalize ----
    // lane n(0-15): column n, K = kq+0..15 with kq=0; lanes 16-31: col n, kq=16.
    int bc  = lane & 15;
    int bkq = (lane >> 4) << 4;                 // 0 or 16
    const float* srow_p = song + (size_t)(srow + bc) * DD + bkq;

    float fv[64];
    float ssq = 0.0f;
    #pragma unroll
    for (int k = 0; k < 4; ++k) {
        const float* p = srow_p + k * 32;
        #pragma unroll
        for (int i = 0; i < 16; ++i) { float x = p[i]; fv[k*16 + i] = x; ssq += x * x; }
    }
    // lane and lane^16 together hold the full 128-dim song row
    ssq += __shfl_xor(ssq, 16, 32);
    float inv = 1.0f / fmaxf(sqrtf(ssq), EPSN);

    v16h Bf[4];
    #pragma unroll
    for (int k = 0; k < 4; ++k) {
        #pragma unroll
        for (int i = 0; i < 16; ++i)
            Bf[k][i] = (_Float16)(fv[k * 16 + i] * inv);
    }

    // ---- A fragment addressing (pred rows, 16x32 MxK layout) ----
    // lanes 0-15: row=lane, halves 0..15 = K(kh..kh+15),
    // kh=0; lanes 16-31: same row, kh=16.
    int ar  = lane & 15;
    int akh = (lane >> 4) << 4;                 // 0 or 16
    bool lo_half = lane < 16;

    // Compile-time-foldable contiguous-range max over accumulator elements.
    auto rmax = [](const v8f& a, int elo, int ehi) -> float {
        float v = -INFINITY;
        #pragma unroll
        for (int e = 0; e < 8; ++e)
            if (e >= elo && e <= ehi) v = fmaxf(v, a[e]);
        return v;
    };

    float cur = -INFINITY;                      // open batch's partial max

    #pragma unroll
    for (int t = 0; t < NTILE; ++t) {
        const int base = 16 * t;
        const _Float16* at = pred_h + (size_t)(base + ar) * DD + akh;

        v8f acc = {};
        #pragma unroll
        for (int k = 0; k < 4; ++k) {
            v16h Af;
            reinterpret_cast<v4f*>(&Af)[0] = *reinterpret_cast<const v4f*>(at + k * 32);
            reinterpret_cast<v4f*>(&Af)[1] = *reinterpret_cast<const v4f*>(at + k * 32 + 8);
            acc = __builtin_amdgcn_wmma_f32_16x16x32_f16(false, Af, false, Bf[k],
                                                         (short)0, acc, false, false);
        }

        // C layout: element e: lanes<16 -> row m=e; lanes>=16 -> row m=e+8.
        // Global pred row = base + m; batch = row / SS.
        const int b0 = base / SS;               // first batch in tile
        const int b1 = (base + 15) / SS;        // last batch in tile (b0 or b0+1)

        // masked max of this tile's rows belonging to b0, per lane-half
        float p0 = rmax(acc, SS * b0 - base,     SS * b0 + SS - 1 - base);
        float p1 = rmax(acc, SS * b0 - base - 8, SS * b0 + SS - 1 - base - 8);
        cur = fmaxf(cur, lo_half ? p0 : p1);

        const bool closes = (SS * (b0 + 1) - 1) <= base + 15;  // compile-time
        if (closes) {
            float mm = fmaxf(cur, __shfl_xor(cur, 16, 32));
            if (lo_half) {
                size_t idx = (size_t)b0 * N + srow + bc;
                sim[idx] = mm * x_inv[idx];
            }
            if (b1 > b0) {
                float q0 = rmax(acc, SS * b1 - base,     SS * b1 + SS - 1 - base);
                float q1 = rmax(acc, SS * b1 - base - 8, SS * b1 + SS - 1 - base - 8);
                cur = lo_half ? q0 : q1;
            } else {
                cur = -INFINITY;
            }
        }
        // note: !closes implies b1 == b0 (a 16-row tile that does not contain
        // the end of b0 cannot contain rows of b0+1)
    }
}

// ---------------------------------------------------------------------------
// Kernel 3: per-batch online-softmax stats for sim and y: (max, sumexp) x2.
// ---------------------------------------------------------------------------
__global__ void __launch_bounds__(256)
stats_kernel(const float* __restrict__ sim, const float* __restrict__ y,
             float* __restrict__ stats, int N) {
    int b = blockIdx.x;
    int t = threadIdx.x;
    const float* sb = sim + (size_t)b * N;
    const float* yb = y   + (size_t)b * N;

    float m1 = -INFINITY, s1 = 0.0f;
    float m2 = -INFINITY, s2 = 0.0f;
    for (int i = t; i < N; i += 256) {
        float x = sb[i];
        if (x > m1) { s1 = s1 * __expf(m1 - x) + 1.0f; m1 = x; }
        else        { s1 += __expf(x - m1); }
        float z = yb[i];
        if (z > m2) { s2 = s2 * __expf(m2 - z) + 1.0f; m2 = z; }
        else        { s2 += __expf(z - m2); }
    }

    __shared__ float rm1[256], rs1[256], rm2[256], rs2[256];
    rm1[t] = m1; rs1[t] = s1; rm2[t] = m2; rs2[t] = s2;
    __syncthreads();
    for (int h = 128; h > 0; h >>= 1) {
        if (t < h) {
            float ma = rm1[t], sa = rs1[t], mb = rm1[t + h], sb2 = rs1[t + h];
            float m = fmaxf(ma, mb);
            rm1[t] = m; rs1[t] = sa * __expf(ma - m) + sb2 * __expf(mb - m);
            ma = rm2[t]; sa = rs2[t]; mb = rm2[t + h]; sb2 = rs2[t + h];
            m = fmaxf(ma, mb);
            rm2[t] = m; rs2[t] = sa * __expf(ma - m) + sb2 * __expf(mb - m);
        }
        __syncthreads();
    }
    if (t == 0) {
        stats[b * 4 + 0] = rm1[0];
        stats[b * 4 + 1] = rs1[0];
        stats[b * 4 + 2] = rm2[0];
        stats[b * 4 + 3] = rs2[0];
    }
}

// ---------------------------------------------------------------------------
// Kernel 4: per-batch loss: loss_b = log(sum_n exp(p_n)) - sum_n t_n * p_n
// where p_n = softmax(sim)_n, t_n = softmax(y)_n.  (Exact rewrite of
// CE(softmax targets, log_softmax(softmax(sim))); p in [0,1] so no LSE shift.)
// ---------------------------------------------------------------------------
__global__ void __launch_bounds__(256)
loss_kernel(const float* __restrict__ sim, const float* __restrict__ y,
            const float* __restrict__ stats, float* __restrict__ lossbuf, int N) {
    int b = blockIdx.x;
    int t = threadIdx.x;
    float m1 = stats[b * 4 + 0];
    float Zs = stats[b * 4 + 1];
    float my = stats[b * 4 + 2];
    float Zy = stats[b * 4 + 3];
    float invZs = 1.0f / Zs;

    const float* sb = sim + (size_t)b * N;
    const float* yb = y   + (size_t)b * N;

    float Se = 0.0f, Sd = 0.0f;
    for (int i = t; i < N; i += 256) {
        float p = __expf(sb[i] - m1) * invZs;
        Se += __expf(p);
        Sd += __expf(yb[i] - my) * p;
    }

    __shared__ float re[256], rd[256];
    re[t] = Se; rd[t] = Sd;
    __syncthreads();
    for (int h = 128; h > 0; h >>= 1) {
        if (t < h) { re[t] += re[t + h]; rd[t] += rd[t + h]; }
        __syncthreads();
    }
    if (t == 0) lossbuf[b] = logf(re[0]) - rd[0] / Zy;
}

// ---------------------------------------------------------------------------
// Kernel 5: finalize — mean over batches, write scalar.
// ---------------------------------------------------------------------------
__global__ void finalize_kernel(const float* __restrict__ lossbuf, float* __restrict__ out) {
    float v = lossbuf[threadIdx.x & 31];
    #pragma unroll
    for (int off = 16; off > 0; off >>= 1)
        v += __shfl_xor(v, off, 32);
    if (threadIdx.x == 0) out[0] = v * (1.0f / (float)BB);
}

// ---------------------------------------------------------------------------
extern "C" void kernel_launch(void* const* d_in, const int* in_sizes, int n_in,
                              void* d_out, int out_size, void* d_ws, size_t ws_size,
                              hipStream_t stream) {
    (void)n_in; (void)out_size; (void)ws_size;
    const float* pred  = (const float*)d_in[0];
    const float* song  = (const float*)d_in[1];
    const float* x_inv = (const float*)d_in[2];
    const float* y     = (const float*)d_in[3];
    float* out = (float*)d_out;

    int N = in_sizes[1] / DD;   // 100000

    char* ws = (char*)d_ws;
    size_t off = 0;
    _Float16* pred_h = (_Float16*)(ws + off); off += (size_t)NROWS * DD * sizeof(_Float16);
    off = (off + 255) & ~(size_t)255;
    float* sim     = (float*)(ws + off);      off += (size_t)BB * N * sizeof(float);
    off = (off + 255) & ~(size_t)255;
    float* stats   = (float*)(ws + off);      off += (size_t)BB * 4 * sizeof(float);
    off = (off + 255) & ~(size_t)255;
    float* lossbuf = (float*)(ws + off);

    // 1) pred normalize (640 waves, packed layout)
    norm_pred_kernel<<<(NROWS * 32 + 255) / 256, 256, 0, stream>>>(pred, pred_h);

    // 2) fused song-normalize + WMMA GEMM + seq-max (zero padding waste)
    int ntiles = N / 16;                       // 6250
    simmax_kernel<<<(ntiles + 7) / 8, 256, 0, stream>>>(song, pred_h, x_inv, sim, N);

    // 3) softmax stats, 4) loss, 5) finalize
    stats_kernel<<<BB, 256, 0, stream>>>(sim, y, stats, N);
    loss_kernel<<<BB, 256, 0, stream>>>(sim, y, stats, lossbuf, N);
    finalize_kernel<<<1, 32, 0, stream>>>(lossbuf, out);
}
